// CausalSelfAttention_54400055771137
// MI455X (gfx1250) — matrix operand order repaired and checked by reading, not compiled, not run on hardware
//
#include <hip/hip_runtime.h>

#define DIM 768
#define HEADS 12
#define SEQ 4096
#define HD 64
#define RMS_EPS 1.1920928955078125e-07f
#define ATTN_SCALE 0.12f

#define USE_ASYNC 1

typedef __attribute__((ext_vector_type(16))) __bf16 v16bf;
typedef __attribute__((ext_vector_type(8)))  __bf16 v8bf;
typedef __attribute__((ext_vector_type(8)))  float  v8f;

union ABf { v16bf v; v8bf h[2]; };

__device__ inline v8f zero8() {
  v8f z;
#pragma unroll
  for (int i = 0; i < 8; ++i) z[i] = 0.0f;
  return z;
}

#if USE_ASYNC
// Async copy 16 bytes global -> LDS (tracked by ASYNCcnt).
__device__ inline void async_cp16(void* lds, const void* gptr) {
  unsigned dst = (unsigned)(unsigned long long)lds;          // LDS byte offset (addr[31:0])
  unsigned long long src = (unsigned long long)gptr;
  asm volatile("global_load_async_to_lds_b128 %0, %1, off"
               :: "v"(dst), "v"(src) : "memory");
}
__device__ inline void async_wait() {
  asm volatile("s_wait_asynccnt 0x0" ::: "memory");
}
#else
__device__ inline void async_cp16(void* lds, const void* gptr) {
  *(v8bf*)lds = *(const v8bf*)gptr;
}
__device__ inline void async_wait() {}
#endif

// ---------------------------------------------------------------- fp32 -> bf16
__global__ void cvt_bf16(const float* __restrict__ in, __bf16* __restrict__ out, int n) {
  int i = blockIdx.x * blockDim.x + threadIdx.x;
  if (i < n) out[i] = (__bf16)in[i];
}

// ------------------------------------------------- QKV GEMM + RMSNorm + rotary
// 4 waves/block; each wave owns a 16(t) x 64(head-dim) tile. The 64x128 bf16
// weight slab is async-staged into double-buffered LDS (copy of slab n+1
// overlaps WMMA on slab n) and shared by all 4 waves.
__global__ __launch_bounds__(128)
void qkv_norm_rope(const __bf16* __restrict__ xb, const __bf16* __restrict__ wb,
                   __bf16* __restrict__ qb, __bf16* __restrict__ kb,
                   __bf16* __restrict__ vt) {
  __shared__ __align__(16) __bf16 Blds[2][64][136];   // 2 x (64 f x 128 k, +8 pad)

  const int tid  = threadIdx.x;
  const int wave = tid >> 5;
  const int lane = tid & 31;
  const int g    = lane >> 4;
  const int lcol = lane & 15;
  const int tbase = (blockIdx.x * 4 + wave) * 16;
  const int head  = blockIdx.y;
  const int mat   = blockIdx.z;   // 0=q 1=k 2=v
  const int fbase = head * HD;

  v8f acc[4];
#pragma unroll
  for (int nt = 0; nt < 4; ++nt) acc[nt] = zero8();

  const __bf16* wbase = wb + (size_t)mat * DIM * DIM;
  const __bf16* arow  = xb + (size_t)(tbase + lcol) * DIM;

  auto stageB = [&](int buf, int kk0) {
    const int f  = tid >> 1;
    const int cb = (tid & 1) * 64;
    const __bf16* srcB = wbase + (size_t)(fbase + f) * DIM + kk0 + cb;
#pragma unroll
    for (int q = 0; q < 8; ++q)
      async_cp16(&Blds[buf][f][cb + q * 8], srcB + q * 8);
  };

  const int nslab = DIM / 128;   // 6
  stageB(0, 0);
  for (int slab = 0; slab < nslab; ++slab) {
    const int cur = slab & 1;
    async_wait();                 // my copies for Blds[cur] complete
    __syncthreads();              // all waves staged cur; all done reading cur^1
    if (slab + 1 < nslab) stageB(cur ^ 1, (slab + 1) * 128);  // overlap next fill

    const int kk0 = slab * 128;
#pragma unroll
    for (int kk2 = 0; kk2 < 128; kk2 += 32) {
      const int ak0 = kk0 + kk2 + g * 16;          // A: 16 contiguous K per half-wave
      ABf a;
      a.h[0] = *(const v8bf*)(arow + ak0);
      a.h[1] = *(const v8bf*)(arow + ak0 + 8);
      const int bk0 = kk2 + g * 16;                // B: 16 contiguous K per half-wave
#pragma unroll
      for (int nt = 0; nt < 4; ++nt) {
        ABf b;
        b.h[0] = *(const v8bf*)(&Blds[cur][nt * 16 + lcol][bk0]);
        b.h[1] = *(const v8bf*)(&Blds[cur][nt * 16 + lcol][bk0 + 8]);
        acc[nt] = __builtin_amdgcn_wmma_f32_16x16x32_bf16(
            false, a.v, false, b.v, (short)0, acc[nt], false, false);
      }
    }
  }

  if (mat < 2) {
    // RMS norm over the 64-wide head dim, then rotary (dims 0-15 with 32-47;
    // dims 16-31 / 48-63 have zero frequency -> identity, per reference).
#pragma unroll
    for (int r = 0; r < 8; ++r) {
      float ss = acc[0][r] * acc[0][r] + acc[1][r] * acc[1][r] +
                 acc[2][r] * acc[2][r] + acc[3][r] * acc[3][r];
      ss += __shfl_xor(ss, 1, 32);
      ss += __shfl_xor(ss, 2, 32);
      ss += __shfl_xor(ss, 4, 32);
      ss += __shfl_xor(ss, 8, 32);
      const float scale = rsqrtf(ss * (1.0f / 64.0f) + RMS_EPS);
#pragma unroll
      for (int nt = 0; nt < 4; ++nt) acc[nt][r] *= scale;

      const int t = tbase + r + 8 * g;
      const float freq = exp2f(-10.0f * (float)lcol / 15.0f);
      const float th = (float)t * freq;
      float sn, cs;
      __sincosf(th, &sn, &cs);
      const float x1 = acc[0][r], x2 = acc[2][r];
      acc[0][r] =  x1 * cs + x2 * sn;
      acc[2][r] = -x1 * sn + x2 * cs;
    }
  }

  if (mat == 2) {
    // store V transposed: [H][D][T]
#pragma unroll
    for (int nt = 0; nt < 4; ++nt)
#pragma unroll
      for (int r = 0; r < 8; ++r) {
        const int t = tbase + r + 8 * g;
        const int d = nt * 16 + lcol;
        vt[((size_t)head * HD + d) * SEQ + t] = (__bf16)acc[nt][r];
      }
  } else {
    __bf16* dst = (mat == 0) ? qb : kb;
#pragma unroll
    for (int nt = 0; nt < 4; ++nt)
#pragma unroll
      for (int r = 0; r < 8; ++r) {
        const int t = tbase + r + 8 * g;
        const int d = nt * 16 + lcol;
        dst[((size_t)head * SEQ + t) * HD + d] = (__bf16)acc[nt][r];
      }
  }
}

// ------------------------------------------------------------ flash attention
// 4 waves/block, each with its own 16-query tile; K/V chunks async-staged into
// double-buffered LDS (fill of chunk n+1 overlaps WMMA/softmax of chunk n).
__global__ __launch_bounds__(128)
void flash_attn(const __bf16* __restrict__ qb, const __bf16* __restrict__ kb,
                const __bf16* __restrict__ vt, __bf16* __restrict__ yb) {
  __shared__ __align__(16) __bf16 Klds[2][32][72];    // 32 keys x 64 d (+8 pad)
  __shared__ __align__(16) __bf16 Vlds[2][64][40];    // 64 d x 32 s (+8 pad)
  __shared__ __align__(16) __bf16 Plds[4][16][40];    // per-wave P tile

  const int tid  = threadIdx.x;
  const int wave = tid >> 5;
  const int lane = tid & 31;
  const int g    = lane >> 4;
  const int lcol = lane & 15;
  const int head  = blockIdx.y;
  const int qbase = (blockIdx.x * 4 + wave) * 16;
  const int qmax  = (blockIdx.x * 4 + 3) * 16;        // largest qbase in block

  // Q in A layout: two 32-wide d-chunks
  ABf qA[2];
  const __bf16* qrow = qb + ((size_t)head * SEQ + (qbase + lcol)) * HD;
#pragma unroll
  for (int dc = 0; dc < 2; ++dc) {
    const int d0 = dc * 32 + g * 16;
    qA[dc].h[0] = *(const v8bf*)(qrow + d0);
    qA[dc].h[1] = *(const v8bf*)(qrow + d0 + 8);
  }

  v8f o[4];
#pragma unroll
  for (int nt = 0; nt < 4; ++nt) o[nt] = zero8();
  float mrow[8], lrow[8];
#pragma unroll
  for (int r = 0; r < 8; ++r) { mrow[r] = -3.0e38f; lrow[r] = 0.0f; }

  auto stageKV = [&](int buf, int kbase) {
    {
      const int row = tid >> 2;
      const int cb  = (tid & 3) * 16;
      const __bf16* srcK = kb + ((size_t)head * SEQ + kbase + row) * HD + cb;
      async_cp16(&Klds[buf][row][cb],     srcK);
      async_cp16(&Klds[buf][row][cb + 8], srcK + 8);
    }
    {
      const int dr = tid >> 1;
      const int cs = (tid & 1) * 16;
      const __bf16* srcV = vt + ((size_t)head * HD + dr) * SEQ + kbase + cs;
      async_cp16(&Vlds[buf][dr][cs],     srcV);
      async_cp16(&Vlds[buf][dr][cs + 8], srcV + 8);
    }
  };

  const int nchunk_blk = (qmax + 47) >> 5;            // uniform across block
  const int nchunk_me  = (qbase + 47) >> 5;           // uniform across wave

  stageKV(0, 0);
  for (int ch = 0; ch < nchunk_blk; ++ch) {
    const int cur = ch & 1;
    const int kbase = ch * 32;
    async_wait();               // my copies for buffer `cur` complete
    __syncthreads();            // all staged; all done reading cur^1
    if (ch + 1 < nchunk_blk) stageKV(cur ^ 1, kbase + 32);  // overlap next fill

    if (ch < nchunk_me) {
      // S = Q * K^T  (16 x 32 tile, contraction over d=64), K from LDS
      v8f sacc[2];
      sacc[0] = zero8(); sacc[1] = zero8();
#pragma unroll
      for (int dc = 0; dc < 2; ++dc) {
        const int bk0 = dc * 32 + g * 16;
#pragma unroll
        for (int nt = 0; nt < 2; ++nt) {
          ABf b;
          b.h[0] = *(const v8bf*)(&Klds[cur][nt * 16 + lcol][bk0]);
          b.h[1] = *(const v8bf*)(&Klds[cur][nt * 16 + lcol][bk0 + 8]);
          sacc[nt] = __builtin_amdgcn_wmma_f32_16x16x32_bf16(
              false, qA[dc].v, false, b.v, (short)0, sacc[nt], false, false);
        }
      }

      // scale + causal mask
      const bool diag = (kbase + 31 > qbase);
#pragma unroll
      for (int nt = 0; nt < 2; ++nt)
#pragma unroll
        for (int r = 0; r < 8; ++r) {
          float sv = sacc[nt][r] * ATTN_SCALE;
          if (diag) {
            const int t  = qbase + r + 8 * g;
            const int sk = kbase + nt * 16 + lcol;
            if (sk > t) sv = -3.0e38f;
          }
          sacc[nt][r] = sv;
        }

      // online softmax; emit P (bf16) to per-wave LDS tile
#pragma unroll
      for (int r = 0; r < 8; ++r) {
        float mx = fmaxf(sacc[0][r], sacc[1][r]);
        mx = fmaxf(mx, __shfl_xor(mx, 1, 32));
        mx = fmaxf(mx, __shfl_xor(mx, 2, 32));
        mx = fmaxf(mx, __shfl_xor(mx, 4, 32));
        mx = fmaxf(mx, __shfl_xor(mx, 8, 32));
        const float mnew = fmaxf(mrow[r], mx);
        const float corr = __expf(mrow[r] - mnew);
        const float p0 = __expf(sacc[0][r] - mnew);
        const float p1 = __expf(sacc[1][r] - mnew);
        float rs = p0 + p1;
        rs += __shfl_xor(rs, 1, 32);
        rs += __shfl_xor(rs, 2, 32);
        rs += __shfl_xor(rs, 4, 32);
        rs += __shfl_xor(rs, 8, 32);
        lrow[r] = lrow[r] * corr + rs;
        mrow[r] = mnew;
#pragma unroll
        for (int nt = 0; nt < 4; ++nt) o[nt][r] *= corr;
        const int prow = r + 8 * g;
        Plds[wave][prow][lcol]      = (__bf16)p0;
        Plds[wave][prow][16 + lcol] = (__bf16)p1;
      }

      // read P back in A layout (DS ops of one wave are in-order)
      ABf pa;
      const int c0 = g * 16;
      pa.h[0] = *(const v8bf*)(&Plds[wave][lcol][c0]);
      pa.h[1] = *(const v8bf*)(&Plds[wave][lcol][c0 + 8]);

      // O += P * V, V from LDS (d-major rows, contiguous along seq)
#pragma unroll
      for (int nt = 0; nt < 4; ++nt) {
        ABf b;
        b.h[0] = *(const v8bf*)(&Vlds[cur][nt * 16 + lcol][g * 16]);
        b.h[1] = *(const v8bf*)(&Vlds[cur][nt * 16 + lcol][g * 16 + 8]);
        o[nt] = __builtin_amdgcn_wmma_f32_16x16x32_bf16(
            false, pa.v, false, b.v, (short)0, o[nt], false, false);
      }
    }
  }

  // normalize and write y as bf16 [T][DIM]
#pragma unroll
  for (int r = 0; r < 8; ++r) {
    const float inv = 1.0f / lrow[r];
    const int t = qbase + r + 8 * g;
#pragma unroll
    for (int nt = 0; nt < 4; ++nt)
      yb[(size_t)t * DIM + head * HD + nt * 16 + lcol] = (__bf16)(o[nt][r] * inv);
  }
}

// ----------------------------------------------------------- output projection
__global__ __launch_bounds__(128)
void out_proj(const __bf16* __restrict__ yb, const __bf16* __restrict__ cpb,
              float* __restrict__ out) {
  __shared__ __align__(16) __bf16 Blds[2][64][136];

  const int tid  = threadIdx.x;
  const int wave = tid >> 5;
  const int lane = tid & 31;
  const int g    = lane >> 4;
  const int lcol = lane & 15;
  const int tbase = (blockIdx.x * 4 + wave) * 16;
  const int fbase = blockIdx.y * 64;

  v8f acc[4];
#pragma unroll
  for (int nt = 0; nt < 4; ++nt) acc[nt] = zero8();

  const __bf16* arow = yb + (size_t)(tbase + lcol) * DIM;

  auto stageB = [&](int buf, int kk0) {
    const int f  = tid >> 1;
    const int cb = (tid & 1) * 64;
    const __bf16* srcB = cpb + (size_t)(fbase + f) * DIM + kk0 + cb;
#pragma unroll
    for (int q = 0; q < 8; ++q)
      async_cp16(&Blds[buf][f][cb + q * 8], srcB + q * 8);
  };

  const int nslab = DIM / 128;   // 6
  stageB(0, 0);
  for (int slab = 0; slab < nslab; ++slab) {
    const int cur = slab & 1;
    async_wait();
    __syncthreads();
    if (slab + 1 < nslab) stageB(cur ^ 1, (slab + 1) * 128);

    const int kk0 = slab * 128;
#pragma unroll
    for (int kk2 = 0; kk2 < 128; kk2 += 32) {
      const int ak0 = kk0 + kk2 + g * 16;
      ABf a;
      a.h[0] = *(const v8bf*)(arow + ak0);
      a.h[1] = *(const v8bf*)(arow + ak0 + 8);
      const int bk0 = kk2 + g * 16;
#pragma unroll
      for (int nt = 0; nt < 4; ++nt) {
        ABf b;
        b.h[0] = *(const v8bf*)(&Blds[cur][nt * 16 + lcol][bk0]);
        b.h[1] = *(const v8bf*)(&Blds[cur][nt * 16 + lcol][bk0 + 8]);
        acc[nt] = __builtin_amdgcn_wmma_f32_16x16x32_bf16(
            false, a.v, false, b.v, (short)0, acc[nt], false, false);
      }
    }
  }

#pragma unroll
  for (int nt = 0; nt < 4; ++nt)
#pragma unroll
    for (int r = 0; r < 8; ++r) {
      const int t = tbase + r + 8 * g;
      out[(size_t)t * DIM + fbase + nt * 16 + lcol] = acc[nt][r];
    }
}

// ------------------------------------------------------------------- launcher
extern "C" void kernel_launch(void* const* d_in, const int* in_sizes, int n_in,
                              void* d_out, int out_size, void* d_ws, size_t ws_size,
                              hipStream_t stream) {
  const float* x        = (const float*)d_in[0];
  const float* qkv_w    = (const float*)d_in[1];
  const float* c_proj_w = (const float*)d_in[2];

  char* ws = (char*)d_ws;
  size_t off = 0;
  auto carve = [&](size_t bytes) -> char* {
    char* p = ws + off;
    off += (bytes + 255) & ~(size_t)255;
    return p;
  };
  __bf16* xb  = (__bf16*)carve((size_t)SEQ * DIM * 2);
  __bf16* wb  = (__bf16*)carve((size_t)3 * DIM * DIM * 2);
  __bf16* cpb = (__bf16*)carve((size_t)DIM * DIM * 2);
  __bf16* qb  = (__bf16*)carve((size_t)HEADS * SEQ * HD * 2);
  __bf16* kb  = (__bf16*)carve((size_t)HEADS * SEQ * HD * 2);
  __bf16* vt  = (__bf16*)carve((size_t)HEADS * SEQ * HD * 2);
  __bf16* yb  = (__bf16*)carve((size_t)SEQ * DIM * 2);

  const int n1 = SEQ * DIM;
  cvt_bf16<<<(n1 + 255) / 256, 256, 0, stream>>>(x, xb, n1);
  const int n2 = 3 * DIM * DIM;
  cvt_bf16<<<(n2 + 255) / 256, 256, 0, stream>>>(qkv_w, wb, n2);
  const int n3 = DIM * DIM;
  cvt_bf16<<<(n3 + 255) / 256, 256, 0, stream>>>(c_proj_w, cpb, n3);

  qkv_norm_rope<<<dim3(SEQ / 64, HEADS, 3), 128, 0, stream>>>(xb, wb, qb, kb, vt);
  flash_attn<<<dim3(SEQ / 64, HEADS), 128, 0, stream>>>(qb, kb, vt, yb);
  out_proj<<<dim3(SEQ / 64, DIM / HD), 128, 0, stream>>>(yb, cpb, (float*)d_out);
}
